// HGPSL_56745107914901
// MI455X (gfx1250) — matrix operand order repaired and checked by reading, not compiled, not run on hardware
//
#include <hip/hip_runtime.h>

// ---------------- problem constants (match reference) ----------------
#define Bd    256
#define Nd    500
#define FEATd 64
#define HIDd  128
#define CLSd  2
#define DEGd  8
#define Egd   (Nd * DEGd)      // 4000 edges per graph
#define Etot  (Bd * Egd)       // 1,024,000 edges
#define NT    (Bd * Nd)        // 128,000 nodes
#define K1d   250
#define K2d   125
#define LAMBf 1.0f
#define NEGSf 0.2f

typedef __attribute__((ext_vector_type(16))) _Float16 v16h;
typedef __attribute__((ext_vector_type(8)))  _Float16 v8h;
typedef __attribute__((ext_vector_type(4)))  _Float16 v4h;
typedef __attribute__((ext_vector_type(8)))  float    v8f;

__device__ __forceinline__ float dinvf(float d) {
    return d > 0.f ? rsqrtf(d) : 0.f;
}

// A-operand fragment (16x32 f16) from LDS row [m][k], k contiguous (32 halfs/row).
// lane 0-15 -> m, K-halves: i<8 -> k=i+hi*8 ; i>=8 -> k=i+8+hi*8 (two runs of 8).
__device__ __forceinline__ v16h load_frag_a(const _Float16* rowp, int hi) {
    v8h lo  = *(const v8h*)(rowp + hi * 8);
    v8h hi8 = *(const v8h*)(rowp + 16 + hi * 8);
    return __builtin_shufflevector(lo, hi8, 0, 1, 2, 3, 4, 5, 6, 7,
                                   8, 9, 10, 11, 12, 13, 14, 15);
}

// B-operand fragment (32x16 f16) from transposed LDS [n][k], k contiguous.
// lane 0-15 -> n, K-halves: i<8 -> k=i+hi*8 ; i>=8 -> k=i+8+hi*8 (two runs of 8).
__device__ __forceinline__ v16h load_frag_b(const _Float16* colp, int hi) {
    v8h lo  = *(const v8h*)(colp + hi * 8);
    v8h hi8 = *(const v8h*)(colp + 16 + hi * 8);
    return __builtin_shufflevector(lo, hi8, 0, 1, 2, 3, 4, 5, 6, 7,
                                   8, 9, 10, 11, 12, 13, 14, 15);
}

// ---------------- simple elementwise kernels ----------------
__global__ void fill_kernel(float* p, float v, int n) {
    int i = blockIdx.x * blockDim.x + threadIdx.x;
    if (i < n) p[i] = v;
}

__global__ void relu_kernel(float* p, int n) {
    int i = blockIdx.x * blockDim.x + threadIdx.x;
    if (i < n) p[i] = fmaxf(p[i], 0.f);
}

// ---------------- sparse GCN stage ----------------
__global__ void edge_deg_kernel(const int* __restrict__ idx, const float* __restrict__ ew,
                                float* __restrict__ deg, int E) {
    int e = blockIdx.x * blockDim.x + threadIdx.x;
    if (e < E) atomicAdd(&deg[idx[e]], ew[e]);
}

__global__ void gcn_base_kernel(const float* __restrict__ hraw, const float* __restrict__ deg,
                                const float* __restrict__ bias, float* __restrict__ out, int n) {
    int i = blockIdx.x * blockDim.x + threadIdx.x;
    if (i >= n) return;
    int v = i / HIDd, f = i % HIDd;
    float di = dinvf(deg[v]);
    out[i] = hraw[i] * di * di + bias[f];
}

__global__ void edge_scatter_kernel(const int* __restrict__ row, const int* __restrict__ col,
                                    const float* __restrict__ ew, const float* __restrict__ deg,
                                    const float* __restrict__ src, float* __restrict__ dst,
                                    float sign) {
    int e = blockIdx.x;
    int rr = row[e], cc = col[e];
    float coef = sign * ew[e] * dinvf(deg[rr]) * dinvf(deg[cc]);
    int f = threadIdx.x;
    atomicAdd(&dst[(size_t)cc * HIDd + f], coef * src[(size_t)rr * HIDd + f]);
}

__global__ void abs_row_sum_kernel(const float* __restrict__ X, float* __restrict__ out,
                                   int rows, int cols) {
    int r = blockIdx.x * blockDim.x + threadIdx.x;
    if (r >= rows) return;
    float s = 0.f;
    for (int c = 0; c < cols; ++c) s += fabsf(X[(size_t)r * cols + c]);
    out[r] = s;
}

// ---------------- per-graph top-k via 512-wide bitonic sort in LDS ----------------
__global__ void topk_kernel(const float* __restrict__ score, int stride, int Ksel,
                            int* __restrict__ idx_out) {
    __shared__ float v[512];
    __shared__ int   ix[512];
    int b = blockIdx.x;
    for (int i = threadIdx.x; i < 512; i += blockDim.x) {
        v[i]  = (i < stride) ? score[(size_t)b * stride + i] : -3.4e38f;
        ix[i] = i;
    }
    __syncthreads();
    for (int k = 2; k <= 512; k <<= 1) {
        for (int j = k >> 1; j > 0; j >>= 1) {
            for (int t = threadIdx.x; t < 512; t += blockDim.x) {
                int p = t ^ j;
                if (p > t) {
                    bool up = ((t & k) == 0);            // descending overall
                    bool sw = up ? (v[t] < v[p]) : (v[t] > v[p]);
                    if (sw) {
                        float tv = v[t]; v[t] = v[p]; v[p] = tv;
                        int   ti = ix[t]; ix[t] = ix[p]; ix[p] = ti;
                    }
                }
            }
            __syncthreads();
        }
    }
    for (int i = threadIdx.x; i < Ksel; i += blockDim.x)
        idx_out[(size_t)b * Ksel + i] = ix[i];
}

__global__ void gather_rows_kernel(const float* __restrict__ src, const int* __restrict__ idx,
                                   float* __restrict__ dst, int Ksel, int srcStride) {
    int g = blockIdx.x;                 // b*Ksel + kk
    int b = g / Ksel;
    int node = idx[g];
    dst[(size_t)g * HIDd + threadIdx.x] =
        src[((size_t)b * srcStride + node) * HIDd + threadIdx.x];
}

__global__ void rank_kernel(const int* __restrict__ idx1, int* __restrict__ rank) {
    int g = blockIdx.x * blockDim.x + threadIdx.x;   // b*K1 + kk
    if (g >= Bd * K1d) return;
    int b = g / K1d, kk = g % K1d;
    rank[b * Nd + idx1[g]] = kk;
}

__global__ void aind_kernel(const int* __restrict__ row, const int* __restrict__ col,
                            const float* __restrict__ ew, const int* __restrict__ rank,
                            float* __restrict__ A_ind, int E) {
    int e = blockIdx.x * blockDim.x + threadIdx.x;
    if (e >= E) return;
    int rr = row[e], cc = col[e];
    int nr = rank[rr], nc = rank[cc];
    if (nr >= 0 && nc >= 0) {
        int b = rr / Nd;
        atomicAdd(&A_ind[((size_t)b * K1d + nr) * K1d + nc], ew[e]);
    }
}

// ---------------- structure learning: s,t projections + row softmax ----------------
__global__ void att_proj_kernel(const float* __restrict__ xp, const float* __restrict__ att,
                                float* __restrict__ s, float* __restrict__ t, int rows) {
    int r = blockIdx.x * blockDim.x + threadIdx.x;
    if (r >= rows) return;
    float sa = 0.f, ta = 0.f;
    for (int h = 0; h < HIDd; ++h) {
        float v = xp[(size_t)r * HIDd + h];
        sa += v * att[h];
        ta += v * att[HIDd + h];
    }
    s[r] = sa; t[r] = ta;
}

__global__ void struct_softmax_kernel(const float* __restrict__ s, const float* __restrict__ t,
                                      float* __restrict__ A, int Kd) {
    __shared__ float red[256];
    int rowg = blockIdx.x;              // b*Kd + i
    int b = rowg / Kd;
    int j = threadIdx.x;
    float myw = 0.f, w = -3.4e38f;
    if (j < Kd) {
        float val = s[rowg] + t[(size_t)b * Kd + j];
        val = val > 0.f ? val : NEGSf * val;
        myw = val + LAMBf * A[(size_t)rowg * Kd + j];
        w = myw;
    }
    red[threadIdx.x] = w; __syncthreads();
    for (int o = 128; o > 0; o >>= 1) {
        if (threadIdx.x < o) red[threadIdx.x] = fmaxf(red[threadIdx.x], red[threadIdx.x + o]);
        __syncthreads();
    }
    float mx = red[0]; __syncthreads();
    float e = (j < Kd) ? __expf(myw - mx) : 0.f;
    red[threadIdx.x] = e; __syncthreads();
    for (int o = 128; o > 0; o >>= 1) {
        if (threadIdx.x < o) red[threadIdx.x] += red[threadIdx.x + o];
        __syncthreads();
    }
    float sum = red[0];
    if (j < Kd) A[(size_t)rowg * Kd + j] = e / sum;
}

__global__ void col_sum_kernel(const float* __restrict__ A, float* __restrict__ deg, int Kd) {
    int g = blockIdx.x * blockDim.x + threadIdx.x;   // b*Kd + j
    if (g >= Bd * Kd) return;
    int b = g / Kd, j = g % Kd;
    const float* Ab = A + (size_t)b * Kd * Kd;
    float s = 0.f;
    for (int i = 0; i < Kd; ++i) s += Ab[(size_t)i * Kd + j];
    deg[g] = s;
}

__global__ void row_sum_offdiag_kernel(const float* __restrict__ A, float* __restrict__ deg, int Kd) {
    int g = blockIdx.x * blockDim.x + threadIdx.x;   // b*Kd + i
    if (g >= Bd * Kd) return;
    int b = g / Kd, i = g % Kd;
    const float* Ar = A + ((size_t)b * Kd + i) * Kd;
    float s = 0.f;
    for (int j = 0; j < Kd; ++j) s += Ar[j];
    deg[g] = s - Ar[i];
}

__global__ void a2_gather_kernel(const float* __restrict__ A1m, const int* __restrict__ idx2,
                                 float* __restrict__ A2m) {
    int g = blockIdx.x * blockDim.x + threadIdx.x;   // b*K2*K2 + i*K2 + j
    if (g >= Bd * K2d * K2d) return;
    int b = g / (K2d * K2d);
    int rem = g % (K2d * K2d);
    int i = rem / K2d, j = rem % K2d;
    int si = idx2[b * K2d + i], sj = idx2[b * K2d + j];
    A2m[g] = A1m[((size_t)b * K1d + si) * K1d + sj];
}

__global__ void readout_kernel(const float* __restrict__ X, float* __restrict__ out, int Kd) {
    int g = blockIdx.x * blockDim.x + threadIdx.x;   // b*HID + f
    if (g >= Bd * HIDd) return;
    int b = g / HIDd, f = g % HIDd;
    const float* Xb = X + (size_t)b * Kd * HIDd + f;
    float mx = -3.4e38f, sm = 0.f;
    for (int k = 0; k < Kd; ++k) {
        float v = Xb[(size_t)k * HIDd];
        mx = fmaxf(mx, v);
        sm += v;
    }
    out[(size_t)b * 2 * HIDd + f]        = mx;
    out[(size_t)b * 2 * HIDd + HIDd + f] = sm / (float)Kd;
}

__global__ void z_combine_kernel(const float* __restrict__ x1, const float* __restrict__ x2,
                                 const float* __restrict__ x3, float* __restrict__ z, int n) {
    int i = blockIdx.x * blockDim.x + threadIdx.x;
    if (i < n) z[i] = fmaxf(x1[i], 0.f) + fmaxf(x2[i], 0.f) + fmaxf(x3[i], 0.f);
}

__global__ void final_kernel(const float* __restrict__ z2, const float* __restrict__ Wl3,
                             const float* __restrict__ bl3, float* __restrict__ out) {
    int b = blockIdx.x * blockDim.x + threadIdx.x;
    if (b >= Bd) return;
    float l0 = bl3[0], l1 = bl3[1];
    for (int k = 0; k < 64; ++k) {
        float v = z2[(size_t)b * 64 + k];
        l0 += v * Wl3[k * CLSd + 0];
        l1 += v * Wl3[k * CLSd + 1];
    }
    float mx  = fmaxf(l0, l1);
    float lse = mx + __logf(__expf(l0 - mx) + __expf(l1 - mx));
    out[b * CLSd + 0] = l0 - lse;
    out[b * CLSd + 1] = l1 - lse;
}

// ---------------- WMMA GEMM, LDS-staged, NO bounds (M%64==N%64==K%32==0) ----------------
// C[M,N] = A[M,K](row-major, lda) * B[K,N](row-major, ldb) [+bias] [ReLU]
// 256 threads = 8 waves; block tile 64x64; wave tile 16x32 (two 16x16 WMMAs).
__global__ __launch_bounds__(256) void wmma_gemm_kernel(
        const float* __restrict__ A, int lda,
        const float* __restrict__ Bm, int ldb,
        const float* __restrict__ bias,
        float* __restrict__ C, int ldc,
        int M, int N, int K, int relu) {
    __shared__ _Float16 ldsA[64 * 32];   // [m][k], k contiguous
    __shared__ _Float16 ldsB[64 * 32];   // [n][k] (transposed), k contiguous
    int m0 = blockIdx.x * 64, n0 = blockIdx.y * 64;
    int tid = threadIdx.x;
    int lane = tid & 31, wave = tid >> 5;
    int wm = wave >> 1, wn = wave & 1;           // 4 (M) x 2 (N) waves
    int r = lane & 15, hi = lane >> 4;
    v8f acc0 = {}, acc1 = {};
    for (int kt = 0; kt < K; kt += 32) {
#pragma unroll
        for (int it = 0; it < 2; ++it) {
            int fid = tid + it * 256;                    // 512 float4 slots each matrix
            // A tile: 64 rows x 32 k  (8 float4 per row)
            int arow = fid >> 3, ac4 = (fid & 7) * 4;
            float4 av = *(const float4*)&A[(size_t)(m0 + arow) * lda + kt + ac4];
            v4h ah = { (_Float16)av.x, (_Float16)av.y, (_Float16)av.z, (_Float16)av.w };
            *(v4h*)&ldsA[arow * 32 + ac4] = ah;
            // B tile: 32 k-rows x 64 n (16 float4 per row), store transposed
            int brow = fid >> 4, bc4 = (fid & 15) * 4;
            float4 bv = *(const float4*)&Bm[(size_t)(kt + brow) * ldb + n0 + bc4];
            ldsB[(bc4 + 0) * 32 + brow] = (_Float16)bv.x;
            ldsB[(bc4 + 1) * 32 + brow] = (_Float16)bv.y;
            ldsB[(bc4 + 2) * 32 + brow] = (_Float16)bv.z;
            ldsB[(bc4 + 3) * 32 + brow] = (_Float16)bv.w;
        }
        __syncthreads();
        v16h a  = load_frag_a(&ldsA[(wm * 16 + r) * 32], hi);
        v16h b0 = load_frag_b(&ldsB[(wn * 32 + r) * 32], hi);
        v16h b1 = load_frag_b(&ldsB[(wn * 32 + 16 + r) * 32], hi);
        acc0 = __builtin_amdgcn_wmma_f32_16x16x32_f16(false, a, false, b0, (short)0, acc0,
                                                      false, false);
        acc1 = __builtin_amdgcn_wmma_f32_16x16x32_f16(false, a, false, b1, (short)0, acc1,
                                                      false, false);
        __syncthreads();
    }
    int c0 = n0 + wn * 32 + r, c1 = c0 + 16;
    float bi0 = bias ? bias[c0] : 0.f;
    float bi1 = bias ? bias[c1] : 0.f;
#pragma unroll
    for (int q = 0; q < 8; ++q) {
        int row = m0 + wm * 16 + q + hi * 8;     // C/D layout: VGPR q -> M=q (+8 lanes 16-31)
        float v0 = acc0[q] + bi0;
        float v1 = acc1[q] + bi1;
        if (relu) { v0 = fmaxf(v0, 0.f); v1 = fmaxf(v1, 0.f); }
        C[(size_t)row * ldc + c0] = v0;
        C[(size_t)row * ldc + c1] = v1;
    }
}

// ---------------- batched normalized-adjacency aggregate (WMMA, LDS-staged) ----------------
// Out[b,j,f] = post( sign * dinv(deg[b,j]) * sum_i dinv(deg[b,i]) * A[b,i,j] * Hm[b,i,f] )
// Bounds handled in staging via clamped addresses + multiplicative masks (no EXEC divergence
// in the WMMA path). Block tile 64(j) x 64(f); wave tile 16x32.
__global__ __launch_bounds__(256) void dense_agg_wmma_kernel(
        const float* __restrict__ A, const float* __restrict__ deg,
        const float* __restrict__ Hm, const float* __restrict__ xin,
        const float* __restrict__ bias, float* __restrict__ Out,
        int Kd, int zero_diag, int relu, float sign) {
    __shared__ _Float16 ldsA[64 * 32];   // [j][i], i contiguous
    __shared__ _Float16 ldsB[64 * 32];   // [f][i], i contiguous
    int b = blockIdx.z;
    int j0 = blockIdx.x * 64, f0 = blockIdx.y * 64;
    const float* Ab   = A   + (size_t)b * Kd * Kd;
    const float* degb = deg + (size_t)b * Kd;
    const float* Hb   = Hm  + (size_t)b * Kd * HIDd;
    int tid = threadIdx.x;
    int lane = tid & 31, wave = tid >> 5;
    int wj = wave >> 1, wf = wave & 1;
    int r = lane & 15, hi = lane >> 4;
    v8f acc0 = {}, acc1 = {};
    int ktiles = (Kd + 31) >> 5;
    for (int kt0 = 0; kt0 < ktiles; ++kt0) {
        int kt = kt0 << 5;
#pragma unroll
        for (int it = 0; it < 2; ++it) {
            int fid = tid + it * 256;
            int irow = fid >> 4;                  // 0..31 : i-local
            int j4 = (fid & 15) * 4;              // 0..60 : j-local / f-local
            int ig = kt + irow;
            int ic = ig < Kd ? ig : Kd - 1;
            float rowmask = (ig < Kd) ? 1.f : 0.f;
            float di = dinvf(degb[ic]) * rowmask;
            // A^T stage: element (m=j, k=i) = dinv[i]*A[i,j], diag dropped if requested
#pragma unroll
            for (int c = 0; c < 4; ++c) {
                int jg = j0 + j4 + c;
                int jc = jg < Kd ? jg : Kd - 1;
                float av = Ab[(size_t)ic * Kd + jc];
                float m = di * ((jg < Kd) ? 1.f : 0.f);
                if (zero_diag && ig == jg) m = 0.f;
                ldsA[(j4 + c) * 32 + irow] = (_Float16)(m * av);
            }
            // H stage (transposed): element (n=f, k=i) = Hm[i,f]
            float4 hv = *(const float4*)&Hb[(size_t)ic * HIDd + f0 + j4];
            ldsB[(j4 + 0) * 32 + irow] = (_Float16)(rowmask * hv.x);
            ldsB[(j4 + 1) * 32 + irow] = (_Float16)(rowmask * hv.y);
            ldsB[(j4 + 2) * 32 + irow] = (_Float16)(rowmask * hv.z);
            ldsB[(j4 + 3) * 32 + irow] = (_Float16)(rowmask * hv.w);
        }
        __syncthreads();
        v16h a  = load_frag_a(&ldsA[(wj * 16 + r) * 32], hi);
        v16h b0 = load_frag_b(&ldsB[(wf * 32 + r) * 32], hi);
        v16h b1 = load_frag_b(&ldsB[(wf * 32 + 16 + r) * 32], hi);
        acc0 = __builtin_amdgcn_wmma_f32_16x16x32_f16(false, a, false, b0, (short)0, acc0,
                                                      false, false);
        acc1 = __builtin_amdgcn_wmma_f32_16x16x32_f16(false, a, false, b1, (short)0, acc1,
                                                      false, false);
        __syncthreads();
    }
    int fc0 = f0 + wf * 32 + r, fc1 = fc0 + 16;
    float bi0 = bias ? bias[fc0] : 0.f;
    float bi1 = bias ? bias[fc1] : 0.f;
#pragma unroll
    for (int q = 0; q < 8; ++q) {
        int j = j0 + wj * 16 + q + hi * 8;
        if (j < Kd) {
            float dj = dinvf(degb[j]);
            float v0 = sign * acc0[q] * dj + bi0;
            float v1 = sign * acc1[q] * dj + bi1;
            if (xin) {
                v0 += xin[((size_t)b * Kd + j) * HIDd + fc0];
                v1 += xin[((size_t)b * Kd + j) * HIDd + fc1];
            }
            if (relu) { v0 = fmaxf(v0, 0.f); v1 = fmaxf(v1, 0.f); }
            Out[((size_t)b * Kd + j) * HIDd + fc0] = v0;
            Out[((size_t)b * Kd + j) * HIDd + fc1] = v1;
        }
    }
}

// ---------------- host side ----------------
#define MiB (1024ull * 1024ull)

extern "C" void kernel_launch(void* const* d_in, const int* in_sizes, int n_in,
                              void* d_out, int out_size, void* d_ws, size_t ws_size,
                              hipStream_t stream) {
    (void)in_sizes; (void)n_in; (void)out_size; (void)ws_size;

    const float* x    = (const float*)d_in[0];
    const float* ea   = (const float*)d_in[1];
    const float* W1   = (const float*)d_in[2];
    const float* b1   = (const float*)d_in[3];
    const float* att1 = (const float*)d_in[4];
    const float* W2   = (const float*)d_in[5];
    const float* b2   = (const float*)d_in[6];
    const float* att2 = (const float*)d_in[7];
    const float* W3   = (const float*)d_in[8];
    const float* b3   = (const float*)d_in[9];
    const float* Wl1  = (const float*)d_in[10];
    const float* bl1  = (const float*)d_in[11];
    const float* Wl2  = (const float*)d_in[12];
    const float* bl2  = (const float*)d_in[13];
    const float* Wl3  = (const float*)d_in[14];
    const float* bl3  = (const float*)d_in[15];
    const int*   ei   = (const int*)d_in[16];
    const int* rowp = ei;
    const int* colp = ei + Etot;

    char* W = (char*)d_ws;
    // three big lifetime-overlaid regions (64 MiB each)
    float* h1raw = (float*)(W + 0);               // NT*HID       (region A)
    float* A1    = (float*)(W + 0);               // B*K1*K1      (region A, after h1raw dead)
    float* hW3   = (float*)(W + 0);               // B*K2*HID     (region A, after A1 dead)
    float* h3    = (float*)(W + 32 * MiB);        // B*K2*HID     (region A)
    float* h1    = (float*)(W + 64 * MiB);        // NT*HID       (region B)
    float* hW    = (float*)(W + 64 * MiB);        // B*K1*HID     (region B, after h1 dead)
    float* A2    = (float*)(W + 64 * MiB);        // B*K2*K2      (region B, after hW dead)
    float* info2 = (float*)(W + 96 * MiB);        // B*K1*HID     (region B upper half)
    float* info  = (float*)(W + 128 * MiB);       // NT*HID       (region C)
    float* xp    = (float*)(W + 128 * MiB);       // B*K1*HID     (region C, after info dead)
    float* xp2   = (float*)(W + 128 * MiB);       // B*K2*HID     (region C, after xp dead)
    float* h2    = (float*)(W + 160 * MiB);       // B*K1*HID     (region C upper half)
    // small buffers
    char* S = W + 192 * MiB;
    float* deg1   = (float*)(S + 0 * MiB);        // NT
    float* deg2   = (float*)(S + 1 * MiB);        // NT
    float* score  = (float*)(S + 2 * MiB);        // NT
    int*   idx1   = (int*)  (S + 3 * MiB);        // B*K1
    int*   rank   = (int*)  (S + 4 * MiB);        // NT
    float* s1     = (float*)(S + 5 * MiB);        // B*K1
    float* t1     = (float*)(S + 6 * MiB);        // B*K1
    float* degD   = (float*)(S + 7 * MiB);        // B*K1
    float* degDr  = (float*)(S + 8 * MiB);        // B*K1
    float* score2 = (float*)(S + 9 * MiB);        // B*K1
    int*   idx2   = (int*)  (S + 10 * MiB);       // B*K2
    float* s2     = (float*)(S + 11 * MiB);       // B*K2
    float* t2     = (float*)(S + 12 * MiB);       // B*K2
    float* degD2  = (float*)(S + 13 * MiB);       // B*K2
    float* x1b    = (float*)(S + 14 * MiB);       // B*2H
    float* x2b    = (float*)(S + 15 * MiB);       // B*2H
    float* x3b    = (float*)(S + 16 * MiB);       // B*2H
    float* zb     = (float*)(S + 17 * MiB);       // B*2H
    float* z1     = (float*)(S + 18 * MiB);       // B*H
    float* z2     = (float*)(S + 19 * MiB);       // B*H/2

    // ---- sparse GCN conv1 ----
    fill_kernel<<<(NT + 255) / 256, 256, 0, stream>>>(deg1, 1.0f, NT);   // self-loop weight
    edge_deg_kernel<<<(Etot + 255) / 256, 256, 0, stream>>>(colp, ea, deg1, Etot);
    {
        dim3 g(NT / 64, HIDd / 64);                              // (2000, 2)
        wmma_gemm_kernel<<<g, 256, 0, stream>>>(x, FEATd, W1, HIDd, nullptr,
                                                h1raw, HIDd, NT, HIDd, FEATd, 0);
    }
    gcn_base_kernel<<<((size_t)NT * HIDd + 255) / 256, 256, 0, stream>>>(h1raw, deg1, b1, h1,
                                                                         NT * HIDd);
    edge_scatter_kernel<<<Etot, HIDd, 0, stream>>>(rowp, colp, ea, deg1, h1raw, h1, 1.0f);
    relu_kernel<<<((size_t)NT * HIDd + 255) / 256, 256, 0, stream>>>(h1, NT * HIDd);

    // ---- node information score (sparse) ----
    hipMemsetAsync(deg2, 0, (size_t)NT * 4, stream);
    edge_deg_kernel<<<(Etot + 255) / 256, 256, 0, stream>>>(rowp, ea, deg2, Etot);
    hipMemcpyAsync(info, h1, (size_t)NT * HIDd * 4, hipMemcpyDeviceToDevice, stream);
    edge_scatter_kernel<<<Etot, HIDd, 0, stream>>>(rowp, colp, ea, deg2, h1, info, -1.0f);
    abs_row_sum_kernel<<<(NT + 255) / 256, 256, 0, stream>>>(info, score, NT, HIDd);

    // ---- pool1: top-K1, gather, induced adjacency ----
    topk_kernel<<<Bd, 256, 0, stream>>>(score, Nd, K1d, idx1);
    gather_rows_kernel<<<Bd * K1d, HIDd, 0, stream>>>(h1, idx1, xp, K1d, Nd);
    hipMemsetAsync(rank, 0xFF, (size_t)NT * 4, stream);
    rank_kernel<<<(Bd * K1d + 255) / 256, 256, 0, stream>>>(idx1, rank);
    hipMemsetAsync(A1, 0, (size_t)Bd * K1d * K1d * 4, stream);
    aind_kernel<<<(Etot + 255) / 256, 256, 0, stream>>>(rowp, colp, ea, rank, A1, Etot);

    // ---- structure learning 1 (softmax in place over A1) + readout 1 ----
    att_proj_kernel<<<(Bd * K1d + 255) / 256, 256, 0, stream>>>(xp, att1, s1, t1, Bd * K1d);
    struct_softmax_kernel<<<Bd * K1d, 256, 0, stream>>>(s1, t1, A1, K1d);
    readout_kernel<<<(Bd * HIDd + 255) / 256, 256, 0, stream>>>(xp, x1b, K1d);

    // ---- dense GCN conv2 ----
    {
        dim3 g(Bd * K1d / 64, HIDd / 64);                        // (1000, 2)
        wmma_gemm_kernel<<<g, 256, 0, stream>>>(xp, HIDd, W2, HIDd, nullptr,
                                                hW, HIDd, Bd * K1d, HIDd, HIDd, 0);
    }
    col_sum_kernel<<<(Bd * K1d + 255) / 256, 256, 0, stream>>>(A1, degD, K1d);
    {
        dim3 g((K1d + 63) / 64, HIDd / 64, Bd);                  // (4, 1... ) -> (4, 2, 256)
        dense_agg_wmma_kernel<<<g, 256, 0, stream>>>(A1, degD, hW, nullptr, b2, h2,
                                                     K1d, 0, 1, 1.0f);
    }

    // ---- node information score (dense) ----
    row_sum_offdiag_kernel<<<(Bd * K1d + 255) / 256, 256, 0, stream>>>(A1, degDr, K1d);
    {
        dim3 g((K1d + 63) / 64, HIDd / 64, Bd);
        dense_agg_wmma_kernel<<<g, 256, 0, stream>>>(A1, degDr, h2, h2, nullptr, info2,
                                                     K1d, 1, 0, -1.0f);
    }
    abs_row_sum_kernel<<<(Bd * K1d + 255) / 256, 256, 0, stream>>>(info2, score2,
                                                                   Bd * K1d, HIDd);

    // ---- pool2: top-K2, gathers ----
    topk_kernel<<<Bd, 256, 0, stream>>>(score2, K1d, K2d, idx2);
    gather_rows_kernel<<<Bd * K2d, HIDd, 0, stream>>>(h2, idx2, xp2, K2d, K1d);
    a2_gather_kernel<<<(Bd * K2d * K2d + 255) / 256, 256, 0, stream>>>(A1, idx2, A2);

    // ---- structure learning 2 + readout 2 ----
    att_proj_kernel<<<(Bd * K2d + 255) / 256, 256, 0, stream>>>(xp2, att2, s2, t2, Bd * K2d);
    struct_softmax_kernel<<<Bd * K2d, 256, 0, stream>>>(s2, t2, A2, K2d);
    readout_kernel<<<(Bd * HIDd + 255) / 256, 256, 0, stream>>>(xp2, x2b, K2d);

    // ---- dense GCN conv3 + readout 3 ----
    {
        dim3 g(Bd * K2d / 64, HIDd / 64);                        // (500, 2)
        wmma_gemm_kernel<<<g, 256, 0, stream>>>(xp2, HIDd, W3, HIDd, nullptr,
                                                hW3, HIDd, Bd * K2d, HIDd, HIDd, 0);
    }
    col_sum_kernel<<<(Bd * K2d + 255) / 256, 256, 0, stream>>>(A2, degD2, K2d);
    {
        dim3 g((K2d + 63) / 64, HIDd / 64, Bd);                  // (2, 2, 256)
        dense_agg_wmma_kernel<<<g, 256, 0, stream>>>(A2, degD2, hW3, nullptr, b3, h3,
                                                     K2d, 0, 1, 1.0f);
    }
    readout_kernel<<<(Bd * HIDd + 255) / 256, 256, 0, stream>>>(h3, x3b, K2d);

    // ---- fuse readouts + MLP head ----
    z_combine_kernel<<<(Bd * 2 * HIDd + 255) / 256, 256, 0, stream>>>(x1b, x2b, x3b, zb,
                                                                      Bd * 2 * HIDd);
    {
        dim3 g(Bd / 64, HIDd / 64);                              // (4, 2)
        wmma_gemm_kernel<<<g, 256, 0, stream>>>(zb, 2 * HIDd, Wl1, HIDd, bl1,
                                                z1, HIDd, Bd, HIDd, 2 * HIDd, 1);
    }
    {
        dim3 g(Bd / 64, (HIDd / 2) / 64);                        // (4, 1)
        wmma_gemm_kernel<<<g, 256, 0, stream>>>(z1, HIDd, Wl2, HIDd / 2, bl2,
                                                z2, HIDd / 2, Bd, HIDd / 2, HIDd, 1);
    }
    final_kernel<<<(Bd + 255) / 256, 256, 0, stream>>>(z2, Wl3, bl3, (float*)d_out);
}
